// _PureSANSeqPredictor_6167573037290
// MI455X (gfx1250) — matrix operand order repaired and checked by reading, not compiled, not run on hardware
//
#include <hip/hip_runtime.h>

typedef __attribute__((ext_vector_type(16))) _Float16 v16h;
typedef __attribute__((ext_vector_type(8)))  _Float16 v8h;
typedef __attribute__((ext_vector_type(8)))  float    v8f;

#define B_    32
#define P_    256
#define W_    16
#define C_    64
#define H_    128
#define PRED_ 64
#define NSEG  4
#define SEGLEN 64   // P_/NSEG

__device__ __forceinline__ v8f wmma16(v16h a, v16h b, v8f c) {
  // D = A(16x32 f16) * B(32x16 f16) + C(16x16 f32)
  return __builtin_amdgcn_wmma_f32_16x16x32_f16(false, a, false, b, (short)0, c, false, false);
}
__device__ __forceinline__ float sigmoidf_(float x) { return 1.0f / (1.0f + expf(-x)); }
__device__ __forceinline__ float gelu_exact(float x) {
  return 0.5f * x * (1.0f + erff(x * 0.70710678118654752f));
}

// Fused GEMM + EMA scan over one P-segment. Branchless A-fragment build:
// every lane issues identical loads; half-wave differences are selects.
template <int IS_STD>
__device__ __forceinline__ void ema_segment(
    const float* mup, const float* stp, const float* rawp,
    float anch, float lsAv, int lo,
    const v16h Bt[4], const float aT[4], v8f acc[4])
{
  v8f zero = {};
  for (int s = 0; s < SEGLEN; ++s) {
    float f0, mval = 0.0f, rsv = 0.0f;
    if (IS_STD) {
      mval = mup[0];
      float sv = stp[0];
      f0  = logf(fmaxf(sv, 1e-3f)) - lsAv;
      rsv = 1.0f / (sv + 1e-5f);
    } else {
      f0 = mup[0] - anch;
    }
    // A fragment: 16x32 f16. lanes 0-15 hold K={0..7,16..23}, lanes 16-31 K={8..15,24..31}
    // K==0 -> f0 ; K in 1..16 -> token(raw[w=K-1]) ; K==17 -> 1.0 (bias row) ; else 0
    const float* rbase = rawp + (lo ? 0 : 8 * C_);
    float r7  = rawp[7  * C_];   // hi lanes' K=8 (w=7); unconditional, always in-bounds
    float r15 = rawp[15 * C_];   // lo lanes' K=16 (w=15); unconditional
    v16h A;
    {
      float t7 = IS_STD ? (r7 - mval) * rsv : (r7 - anch);
      A[0] = lo ? (_Float16)f0 : (_Float16)t7;
    }
    #pragma unroll
    for (int e = 1; e < 8; ++e) {          // lo: w=e-1 ; hi: w=e+7
      float r = rbase[(e - 1) * C_];
      float t = IS_STD ? (r - mval) * rsv : (r - anch);
      A[e] = (_Float16)t;
    }
    {
      float t15 = IS_STD ? (r15 - mval) * rsv : (r15 - anch);
      A[8] = lo ? (_Float16)t15 : (_Float16)0.0f;
    }
    A[9] = lo ? (_Float16)1.0f : (_Float16)0.0f;   // bias row K=17
    #pragma unroll
    for (int e = 10; e < 16; ++e) A[e] = (_Float16)0.0f;

    #pragma unroll
    for (int t = 0; t < 4; ++t) {
      v8f u = wmma16(A, Bt[t], zero);                // u' = (1-a) ⊙ (token@W + b)
      #pragma unroll
      for (int j = 0; j < 8; ++j)
        acc[t][j] = fmaf(aT[t], acc[t][j], u[j]);    // h = a⊙h + u'
    }
    mup += C_; stp += C_; rawp += W_ * C_;
  }
}

// One block = one branch (blockIdx.x>>7) x 16 bc-sequences. 8 waves = 4 P-segments x 2 N-halves.
__global__ __launch_bounds__(256) void fused_kernel(
    const float* __restrict__ mu_hist, const float* __restrict__ std_hist,
    const float* __restrict__ anchor,  const float* __restrict__ raw,
    const float* __restrict__ m_inW, const float* __restrict__ m_inb,
    const float* __restrict__ m_a,   const float* __restrict__ m_pW,
    const float* __restrict__ m_pb,  const float* __restrict__ m_oW,
    const float* __restrict__ m_ob,  const float* __restrict__ m_g,
    const float* __restrict__ s_inW, const float* __restrict__ s_inb,
    const float* __restrict__ s_a,   const float* __restrict__ s_pW,
    const float* __restrict__ s_pb,  const float* __restrict__ s_oW,
    const float* __restrict__ s_ob,  const float* __restrict__ s_g,
    float* __restrict__ out)
{
  const int isStd = (int)(blockIdx.x >> 7);
  const int bx    = (int)(blockIdx.x & 127);

  // uniform per-branch parameter selection
  const float* in_W    = isStd ? s_inW : m_inW;
  const float* in_b    = isStd ? s_inb : m_inb;
  const float* a_logit = isStd ? s_a   : m_a;
  const float* post_W  = isStd ? s_pW  : m_pW;
  const float* post_b  = isStd ? s_pb  : m_pb;
  const float* out_W   = isStd ? s_oW  : m_oW;
  const float* out_b   = isStd ? s_ob  : m_ob;
  const float* gamma   = isStd ? s_g   : m_g;
  float* outp = out + (size_t)isStd * ((size_t)B_ * PRED_ * C_);

  __shared__ float sPart[NSEG * 16 * H_];                // 32 KB; seg-0 reused as hbuf
  __shared__ __align__(16) _Float16 gbuf[16 * H_];       // 4 KB
  __shared__ __align__(16) _Float16 h2buf[16 * H_];      // 4 KB
  __shared__ float lsRed[16 * 16];
  __shared__ float lsAbuf[16];

  const int tid   = threadIdx.x;
  const int lane  = tid & 31;
  const int wave  = tid >> 5;
  const int seg   = wave >> 1;
  const int nbase = (wave & 1) * 64;
  const int lo    = (lane < 16) ? 1 : 0;
  const int m16   = lane & 15;

  const int b  = bx >> 2;
  const int c0 = (bx & 3) * 16;
  const int c  = c0 + m16;           // this lane's channel (A-row M = m16)
  const int bc = b * C_ + c;

  // ---------- Stage 0 (std blocks): block-local logstd anchor over P ----------
  if (isStd) {
    int cl = tid & 15, sg = tid >> 4;
    const float* sp = std_hist + ((size_t)b * P_ + sg * 16) * C_ + c0 + cl;
    float s = 0.0f;
    #pragma unroll
    for (int t = 0; t < 16; ++t) s += logf(fmaxf(sp[t * C_], 1e-3f));
    lsRed[sg * 16 + cl] = s;
  }
  __syncthreads();
  if (isStd && tid < 16) {
    float s = 0.0f;
    #pragma unroll
    for (int g = 0; g < 16; ++g) s += lsRed[g * 16 + tid];
    lsAbuf[tid] = s * (1.0f / (float)P_);
  }
  __syncthreads();

  // ---------- Stage 1: segment EMA partials via fused GEMM+scan ----------
  // B'[k][n] = in_W[k][n]*(1-a[n]) for k<17 ; in_b[n]*(1-a[n]) for k==17 ; 0 pad
  v16h Bt[4];
  float aT[4];
  #pragma unroll
  for (int t = 0; t < 4; ++t) {
    int n = nbase + 16 * t + m16;
    float a   = sigmoidf_(a_logit[n]);
    float oma = 1.0f - a;
    aT[t] = a;
    v16h bb;
    #pragma unroll
    for (int e = 0; e < 16; ++e) {
      int K = (lo ? 0 : 8) + (e < 8 ? e : e + 8);   // B 32x16 f16 layout (table K order, matches A)
      float v = (K < 17)  ? in_W[K * H_ + n] * oma
              : (K == 17) ? in_b[n] * oma
                          : 0.0f;
      bb[e] = (_Float16)v;
    }
    Bt[t] = bb;
  }

  v8f zero = {};
  v8f acc[4];
  #pragma unroll
  for (int t = 0; t < 4; ++t) acc[t] = zero;

  const float anch = anchor[bc];
  const float lsAv = isStd ? lsAbuf[m16] : 0.0f;

  const int p0 = seg * SEGLEN;
  const float* mup  = mu_hist  + ((size_t)b * P_ + p0) * C_ + c;
  const float* stp  = std_hist + ((size_t)b * P_ + p0) * C_ + c;
  const float* rawp = raw + (((size_t)b * P_ + p0) * W_) * C_ + c;

  if (isStd) ema_segment<1>(mup, stp, rawp, anch, lsAv, lo, Bt, aT, acc);
  else       ema_segment<0>(mup, stp, rawp, anch, lsAv, lo, Bt, aT, acc);

  // store segment partials (D layout: row M = lo? j : j+8, col n)
  #pragma unroll
  for (int t = 0; t < 4; ++t) {
    int n = nbase + 16 * t + m16;
    #pragma unroll
    for (int j = 0; j < 8; ++j) {
      int M = lo ? j : j + 8;
      sPart[(seg * 16 + M) * H_ + n] = acc[t][j];
    }
  }
  __syncthreads();

  // ---------- Stage 2: combine segments (h = a^64 * h + part_s), gelu ----------
  float* hbuf = sPart;   // aliases seg-0 region; each (M,n) touched by one thread
  #pragma unroll
  for (int r = 0; r < 8; ++r) {
    int idx = r * 256 + tid;     // 0..2047 over (M,n)
    int M = idx >> 7;
    int n = idx & (H_ - 1);
    float a  = sigmoidf_(a_logit[n]);
    float ap = a * a; ap *= ap; ap *= ap; ap *= ap; ap *= ap; ap *= ap;  // a^64
    float h = 0.0f;
    #pragma unroll
    for (int s = 0; s < NSEG; ++s) h = fmaf(ap, h, sPart[(s * 16 + M) * H_ + n]);
    hbuf[M * H_ + n] = h;
    gbuf[M * H_ + n] = (_Float16)gelu_exact(h);
  }
  __syncthreads();

  // ---------- Stage 3: h2 = h + gelu(h) @ post_W + post_b ----------
  {
    int n = 16 * wave + m16;                    // each wave owns one 16-col N tile
    v8f c3;
    float pb = post_b[n];
    #pragma unroll
    for (int j = 0; j < 8; ++j) { int M = lo ? j : j + 8; c3[j] = hbuf[M * H_ + n] + pb; }
    #pragma unroll
    for (int kc = 0; kc < 4; ++kc) {
      const _Float16* rowp = &gbuf[m16 * H_ + 32 * kc + (lo ? 0 : 8)];
      v8h x0 = *(const v8h*)rowp;               // K base .. base+7
      v8h x1 = *(const v8h*)(rowp + 16);        // K base+16 .. base+23
      v16h A;
      #pragma unroll
      for (int i = 0; i < 8; ++i) { A[i] = x0[i]; A[8 + i] = x1[i]; }
      v16h Bf;
      #pragma unroll
      for (int e = 0; e < 16; ++e) {
        int K = 32 * kc + (lo ? 0 : 8) + (e < 8 ? e : e + 8);
        Bf[e] = (_Float16)post_W[K * H_ + n];
      }
      c3 = wmma16(A, Bf, c3);
    }
    #pragma unroll
    for (int j = 0; j < 8; ++j) { int M = lo ? j : j + 8; h2buf[M * H_ + n] = (_Float16)c3[j]; }
  }
  __syncthreads();

  // ---------- Stage 4: res = h2 @ out_W + out_b, epilogue ----------
  if (wave < 4) {                               // wave-uniform: EXEC stays full
    int n = 16 * wave + m16;                    // pred index
    v8f c4;
    float ob = out_b[n];
    #pragma unroll
    for (int j = 0; j < 8; ++j) c4[j] = ob;
    #pragma unroll
    for (int kc = 0; kc < 4; ++kc) {
      const _Float16* rowp = &h2buf[m16 * H_ + 32 * kc + (lo ? 0 : 8)];
      v8h x0 = *(const v8h*)rowp;
      v8h x1 = *(const v8h*)(rowp + 16);
      v16h A;
      #pragma unroll
      for (int i = 0; i < 8; ++i) { A[i] = x0[i]; A[8 + i] = x1[i]; }
      v16h Bf;
      #pragma unroll
      for (int e = 0; e < 16; ++e) {
        int K = 32 * kc + (lo ? 0 : 8) + (e < 8 ? e : e + 8);
        Bf[e] = (_Float16)out_W[K * PRED_ + n];
      }
      c4 = wmma16(A, Bf, c4);
    }
    #pragma unroll
    for (int j = 0; j < 8; ++j) {
      int M   = lo ? j : j + 8;
      int cc  = c0 + M;
      int bcc = b * C_ + cc;
      float r = c4[j];
      float o;
      if (isStd) o = fmaxf(expf(lsAbuf[M] + gamma[cc] * r), 1e-3f);
      else       o = anchor[bcc] + gamma[cc] * r;
      outp[((size_t)b * PRED_ + n) * C_ + cc] = o;   // (B, PRED, C)
    }
  }
}

extern "C" void kernel_launch(void* const* d_in, const int* in_sizes, int n_in,
                              void* d_out, int out_size, void* d_ws, size_t ws_size,
                              hipStream_t stream) {
  (void)in_sizes; (void)n_in; (void)out_size; (void)d_ws; (void)ws_size;
  const float* mu_hist   = (const float*)d_in[0];
  const float* std_hist  = (const float*)d_in[1];
  const float* anchor    = (const float*)d_in[2];
  const float* raw       = (const float*)d_in[3];
  const float* mean_in_W = (const float*)d_in[4];
  const float* mean_in_b = (const float*)d_in[5];
  const float* mean_a    = (const float*)d_in[6];
  const float* mean_pW   = (const float*)d_in[7];
  const float* mean_pb   = (const float*)d_in[8];
  const float* mean_oW   = (const float*)d_in[9];
  const float* mean_ob   = (const float*)d_in[10];
  const float* gamma_mu  = (const float*)d_in[11];
  const float* std_in_W  = (const float*)d_in[12];
  const float* std_in_b  = (const float*)d_in[13];
  const float* std_a     = (const float*)d_in[14];
  const float* std_pW    = (const float*)d_in[15];
  const float* std_pb    = (const float*)d_in[16];
  const float* std_oW    = (const float*)d_in[17];
  const float* std_ob    = (const float*)d_in[18];
  const float* gamma_std = (const float*)d_in[19];
  float* out = (float*)d_out;

  // single launch: blocks 0-127 mean branch, 128-255 std branch
  fused_kernel<<<256, 256, 0, stream>>>(
      mu_hist, std_hist, anchor, raw,
      mean_in_W, mean_in_b, mean_a, mean_pW, mean_pb, mean_oW, mean_ob, gamma_mu,
      std_in_W, std_in_b, std_a, std_pW, std_pb, std_oW, std_ob, gamma_std,
      out);
}
